// SurvLatentODE_82935818485816
// MI455X (gfx1250) — compile-verified, matrix operand order checked by reading, not run on hardware
//
#include <hip/hip_runtime.h>
#include <hip/hip_bf16.h>

#define Bv 512
#define Tv 64
#define NLv 32
#define LATv 64
#define HIDv 128
#define SCALEF 365.0f
#define STEPF 0.05f
#define MAXG 128

typedef __attribute__((ext_vector_type(16))) _Float16 v16h;
typedef __attribute__((ext_vector_type(8)))  float    v8f;

// ---------------------------------------------------------------------------
// WMMA helpers (CDNA5: V_WMMA_F32_16X16X32_F16, wave32)
// ---------------------------------------------------------------------------
__device__ inline v8f wmma16(v16h a, v16h b, v8f c) {
  return __builtin_amdgcn_wmma_f32_16x16x32_f16(false, a, false, b, (short)0, c,
                                                false, false);
}

// K-offset inside a 32-wide K chunk for element e (0..15) of a v16h fragment.
// Per ISA 7.12.2 (16-bit A 16x32): lanes 0-15: VGPR0..3 -> K 0..7, VGPR4..7 ->
// K16..23; lanes 16-31: +8.  e<8 -> k = half*8 + e ; e>=8 -> k = 16+half*8+(e-8)
__device__ inline int frag_k(int e, int half) {
  return ((e < 8) ? 0 : 16) + half * 8 + (e & 7);
}

// Build A fragment (16x32 tile) from row-major f32 matrix.
__device__ inline v16h load_a_frag_f(const float* A, int lda, int mbase,
                                     int kbase, int lane) {
  const float* row =
      A + (size_t)(mbase + (lane & 15)) * lda + kbase + (lane >> 4) * 8;
  v16h a;
#pragma unroll
  for (int e = 0; e < 8; ++e) a[e] = (_Float16)row[e];
#pragma unroll
  for (int e = 0; e < 8; ++e) a[8 + e] = (_Float16)row[16 + e];
  return a;
}

// Build A fragment from row-major f16 matrix (typically resident in LDS).
__device__ inline v16h load_a_frag_h(const _Float16* A, int lda, int mbase,
                                     int kbase, int lane) {
  const _Float16* row =
      A + (size_t)(mbase + (lane & 15)) * lda + kbase + (lane >> 4) * 8;
  v16h a;
#pragma unroll
  for (int e = 0; e < 8; ++e) a[e] = row[e];
#pragma unroll
  for (int e = 0; e < 8; ++e) a[8 + e] = row[16 + e];
  return a;
}

// ---------------------------------------------------------------------------
// Async global -> LDS staging (GLOBAL_LOAD_ASYNC_TO_LDS_B128, ASYNCcnt).
// Builtin signature (from hipcc diagnostic): params are pointers to
// vector_size(16) int, global source in AS1 ("__device__"), LDS dest in AS3.
// Guarded: falls back to a plain 16-byte copy if the builtins are absent.
// ---------------------------------------------------------------------------
#if __has_builtin(__builtin_amdgcn_global_load_async_to_lds_b128) && \
    __has_builtin(__builtin_amdgcn_s_wait_asynccnt)
#define ASYNC_LDS_OK 1
#endif

typedef int b128_t __attribute__((vector_size(16)));

__device__ inline void stage16(const void* g, void* l) {
#if defined(ASYNC_LDS_OK)
  // AS1 global pointer (64-bit) / AS3 LDS pointer (low 32 bits = LDS offset)
  __builtin_amdgcn_global_load_async_to_lds_b128(
      (__attribute__((address_space(1))) b128_t*)(unsigned long long)(uintptr_t)g,
      (__attribute__((address_space(3))) b128_t*)(unsigned int)(uintptr_t)l,
      0, 0);
#else
  *(int4*)l = *(const int4*)g;
#endif
}

__device__ inline void stage_wait() {
#if defined(ASYNC_LDS_OK)
  __builtin_amdgcn_s_wait_asynccnt(0);
#endif
}

// ---------------------------------------------------------------------------
// Weight packing: f32 [K x N] -> per-fragment, per-lane B layout
// frag index = nt*KT + kt ; each fragment = 32 lanes * 16 halfs (contiguous)
// ---------------------------------------------------------------------------
__global__ void k_pack(const float* __restrict__ W, int K, int N, int KT,
                       int NT, _Float16* __restrict__ out) {
  int idx = blockIdx.x * blockDim.x + threadIdx.x;
  int total = NT * KT * 32;
  if (idx >= total) return;
  int lane = idx & 31;
  int frag = idx >> 5;
  int kt = frag % KT;
  int nt = frag / KT;
  int n = nt * 16 + (lane & 15);
  int half = lane >> 4;
  _Float16* dst = out + (size_t)frag * 512 + lane * 16;
#pragma unroll
  for (int e = 0; e < 16; ++e) {
    int k = kt * 32 + frag_k(e, half);
    float v = (k < K && n < N) ? W[(size_t)k * N + n] : 0.f;
    dst[e] = (_Float16)v;
  }
}

// ---------------------------------------------------------------------------
// Generic WMMA GEMM: C[M x N] = act(A[M x Kp] @ Bpacked + bias)
// Each wave computes a 16 x (16*WN) strip: the A fragment is loaded once per
// K-chunk and reused for WN WMMAs (cuts A traffic WN-fold).
// Kp multiple of 32 (A physically zero-padded), M mult of 16, NT mult of WN.
// act: 0=none, 1=tanh, 2=relu
// ---------------------------------------------------------------------------
template <int WN>
__global__ void k_gemm(const float* __restrict__ A, int lda,
                       const _Float16* __restrict__ Bp,
                       const float* __restrict__ bias, float* __restrict__ C,
                       int ldc, int M, int Kp, int N, int act) {
  int gw = (blockIdx.x * blockDim.x + threadIdx.x) >> 5;
  int lane = threadIdx.x & 31;
  int NT = N >> 4;
  int KT = Kp >> 5;
  int NG = NT / WN;
  int mt = gw / NG;
  int ng = gw - mt * NG;
  if (mt * 16 >= M) return;
  int nt0 = ng * WN;
  const v8f vzero = {0.f, 0.f, 0.f, 0.f, 0.f, 0.f, 0.f, 0.f};
  v8f acc[WN];
#pragma unroll
  for (int w = 0; w < WN; ++w) acc[w] = vzero;
  for (int kt = 0; kt < KT; ++kt) {
    v16h a = load_a_frag_f(A, lda, mt * 16, kt * 32, lane);
#pragma unroll
    for (int w = 0; w < WN; ++w) {
      v16h b =
          *(const v16h*)(Bp + ((size_t)((nt0 + w) * KT + kt)) * 512 + lane * 16);
      acc[w] = wmma16(a, b, acc[w]);
    }
  }
  int mb = mt * 16 + 8 * (lane >> 4);
#pragma unroll
  for (int w = 0; w < WN; ++w) {
    int n = (nt0 + w) * 16 + (lane & 15);
    float bvv = bias ? bias[n] : 0.f;
#pragma unroll
    for (int r = 0; r < 8; ++r) {
      float v = acc[w][r] + bvv;
      if (act == 1) v = tanhf(v);
      else if (act == 2) v = fmaxf(v, 0.f);
      C[(size_t)(mb + r) * ldc + n] = v;
    }
  }
}

// ---------------------------------------------------------------------------
// Setup kernels
// ---------------------------------------------------------------------------
__global__ void k_init(int* ipar, float* acc) {
  if (threadIdx.x == 0) {
    ipar[0] = 2147483647;        // tmin
    ipar[1] = -2147483647 - 1;   // tmax
    ipar[2] = 1;                 // niters
    acc[0] = 0.f;                // recon sq sum
    acc[1] = 0.f;                // n_obs
    acc[2] = 0.f;                // kl sum
  }
}

__global__ void k_minmax(const int* __restrict__ tp, int* ipar) {
  int idx = blockIdx.x * blockDim.x + threadIdx.x;
  if (idx >= Bv * Tv) return;
  atomicMin(&ipar[0], tp[idx]);
  atomicMax(&ipar[1], tp[idx]);
}

__global__ void k_grid(int* ipar, float* fpar, float* h_arr) {
  __shared__ float st0, st1;
  __shared__ int sni;
  if (threadIdx.x == 0) {
    float t0 = (float)ipar[0] / SCALEF;
    float t1 = (float)ipar[1] / SCALEF;
    int ni = (int)ceilf((t1 - t0) / STEPF + 1.0f);
    if (ni < 1) ni = 1;
    if (ni > MAXG) ni = MAXG;
    ipar[2] = ni;
    fpar[0] = t0;
    fpar[1] = t1;
    st0 = t0; st1 = t1; sni = ni;
  }
  __syncthreads();
  int i = threadIdx.x;
  if (i < MAXG) {
    float h = 0.f;
    if (i < sni - 1) {
      float gl = st0 + i * STEPF;
      float gr = st0 + (i + 1) * STEPF;
      if (i + 1 == sni - 1 && gr > st1) gr = st1;  // clamp last grid point
      h = gr - gl;
    }
    h_arr[i] = h;
  }
}

// Build reversed GRU input sequence, zero-padded to K=96:
// Xrev[s][b][0..95] = [obs*mask (32), mask (32), dt (1), zeros]
__global__ void k_build_x(const float* __restrict__ obs,
                          const unsigned char* __restrict__ mask,
                          const int* __restrict__ tp, float* __restrict__ Xrev) {
  int idx = blockIdx.x * blockDim.x + threadIdx.x;
  const int total = Tv * Bv * 96;
  if (idx >= total) return;
  int c = idx % 96;
  int b = (idx / 96) % Bv;
  int s = idx / (96 * Bv);
  int t = Tv - 1 - s;  // reversed sequence
  float v = 0.f;
  if (c < 32) {
    int o = (b * Tv + t) * NLv + c;
    v = mask[o] ? obs[o] : 0.f;
  } else if (c < 64) {
    v = mask[(b * Tv + t) * NLv + (c - 32)] ? 1.f : 0.f;
  } else if (c == 64) {
    v = (t == 0) ? 0.f : (float)(tp[b * Tv + t] - tp[b * Tv + t - 1]);
  }
  Xrev[idx] = v;
}

// ---------------------------------------------------------------------------
// Persistent GRU: one launch, 16 blocks x 32 batch rows, 64 recurrent steps.
// Whh fragments async-staged into LDS; h kept in LDS as f16; gi precomputed.
// Per K-chunk one LDS A-fragment load feeds 6 WMMAs.
// ---------------------------------------------------------------------------
__global__ void __launch_bounds__(256) k_gru(const float* __restrict__ gi_all,
                                             const _Float16* __restrict__ Whh_p,
                                             const float* __restrict__ bhh,
                                             float* __restrict__ h_final) {
  extern __shared__ char smem[];
  _Float16* sW = (_Float16*)smem;       // 24*4*512 = 49152 halfs (96 KB)
  _Float16* sh = sW + 49152;            // 32*128 halfs
  float* sgh = (float*)(sh + 32 * 128); // 32*384 floats
  int tid = threadIdx.x, lane = tid & 31, wave = tid >> 5;
  int row0 = blockIdx.x * 32;
  for (int i = tid * 8; i < 49152; i += 256 * 8) stage16(Whh_p + i, sW + i);
  for (int i = tid; i < 32 * 128; i += 256) sh[i] = (_Float16)0.f;
  stage_wait();
  __syncthreads();
  int mt = wave >> 2;
  int nt0 = (wave & 3) * 6;
  const v8f vzero = {0.f, 0.f, 0.f, 0.f, 0.f, 0.f, 0.f, 0.f};
  for (int t = 0; t < Tv; ++t) {
    // gh = h @ Whh : 2x24 tiles over 8 waves (6 each), K = 128
    v8f acc[6];
#pragma unroll
    for (int j = 0; j < 6; ++j) acc[j] = vzero;
    for (int kt = 0; kt < 4; ++kt) {
      v16h a = load_a_frag_h(sh, 128, mt * 16, kt * 32, lane);
#pragma unroll
      for (int j = 0; j < 6; ++j) {
        v16h b =
            *(const v16h*)(sW + (size_t)((nt0 + j) * 4 + kt) * 512 + lane * 16);
        acc[j] = wmma16(a, b, acc[j]);
      }
    }
    int mb = mt * 16 + 8 * (lane >> 4);
#pragma unroll
    for (int j = 0; j < 6; ++j) {
      int n = (nt0 + j) * 16 + (lane & 15);
#pragma unroll
      for (int r = 0; r < 8; ++r) sgh[(mb + r) * 384 + n] = acc[j][r];
    }
    __syncthreads();
    // gate update
    const float* gi = gi_all + ((size_t)t * Bv + row0) * 384;
    for (int i = tid; i < 32 * 128; i += 256) {
      int m = i >> 7, j2 = i & 127;
      float ghr = sgh[m * 384 + j2] + bhh[j2];
      float ghz = sgh[m * 384 + 128 + j2] + bhh[128 + j2];
      float ghn = sgh[m * 384 + 256 + j2] + bhh[256 + j2];
      float r = 1.f / (1.f + __expf(-(gi[m * 384 + j2] + ghr)));
      float zg = 1.f / (1.f + __expf(-(gi[m * 384 + 128 + j2] + ghz)));
      float nn = tanhf(gi[m * 384 + 256 + j2] + r * ghn);
      float hold = (float)sh[i];
      sh[i] = (_Float16)((1.f - zg) * nn + zg * hold);
    }
    __syncthreads();
  }
  for (int i = tid; i < 32 * 128; i += 256)
    h_final[(size_t)(row0 + (i >> 7)) * 128 + (i & 127)] = (float)sh[i];
}

// ---------------------------------------------------------------------------
// z0 = mu + eps*exp(0.5*logvar) -> ys[0]; accumulate KL term.
// ---------------------------------------------------------------------------
__global__ void k_z0(const float* __restrict__ mu, const float* __restrict__ lv,
                     const float* __restrict__ eps, float* __restrict__ ys,
                     float* acc) {
  __shared__ float red[256];
  int idx = blockIdx.x * blockDim.x + threadIdx.x;
  float term = 0.f;
  if (idx < Bv * LATv) {
    float m = mu[idx], l = lv[idx];
    ys[idx] = m + eps[idx] * __expf(0.5f * l);
    term = 1.f + l - m * m - __expf(l);
  }
  red[threadIdx.x] = term;
  __syncthreads();
  for (int s = 128; s > 0; s >>= 1) {
    if ((int)threadIdx.x < s) red[threadIdx.x] += red[threadIdx.x + s];
    __syncthreads();
  }
  if (threadIdx.x == 0) atomicAdd(acc + 2, red[0]);
}

// ---------------------------------------------------------------------------
// Persistent RK4 ODE integrator: one launch, 16 blocks x 32 rows, all steps.
// W1/W2/W3 fragments + activations + z + k1..k4 live in LDS; every layer is
// a WMMA GEMM; trajectory rows stream to ys for later interpolation.
// ---------------------------------------------------------------------------
__global__ void __launch_bounds__(256) k_ode(
    const _Float16* __restrict__ W1p, const _Float16* __restrict__ W2p,
    const _Float16* __restrict__ W3p, const float* __restrict__ b1,
    const float* __restrict__ b2, const float* __restrict__ b3,
    const int* __restrict__ ipar, const float* __restrict__ h_arr,
    float* __restrict__ ys) {
  extern __shared__ char smem[];
  _Float16* sW1 = (_Float16*)smem;   // 8192 halfs
  _Float16* sW2 = sW1 + 8192;        // 16384 halfs
  _Float16* sW3 = sW2 + 16384;       // 8192 halfs
  _Float16* szin = sW3 + 8192;       // 32*64
  _Float16* sa1 = szin + 2048;       // 32*128
  _Float16* sa2 = sa1 + 4096;        // 32*128
  float* sz = (float*)(sa2 + 4096);  // 32*64 floats
  float* sk = sz + 2048;             // 4 * 32*64 floats
  int tid = threadIdx.x, lane = tid & 31, wave = tid >> 5;
  int row0 = blockIdx.x * 32;
  for (int i = tid * 8; i < 8192; i += 256 * 8) stage16(W1p + i, sW1 + i);
  for (int i = tid * 8; i < 16384; i += 256 * 8) stage16(W2p + i, sW2 + i);
  for (int i = tid * 8; i < 8192; i += 256 * 8) stage16(W3p + i, sW3 + i);
  {
    const float* zrow = ys + (size_t)row0 * 64;  // ys[0], rows contiguous
    for (int i = tid * 4; i < 2048; i += 256 * 4) stage16(zrow + i, sz + i);
  }
  stage_wait();
  __syncthreads();
  int ni = ipar[2];
  int mt = wave >> 2;
  const v8f vzero = {0.f, 0.f, 0.f, 0.f, 0.f, 0.f, 0.f, 0.f};
  for (int step = 0; step < MAXG - 1; ++step) {
    if (step >= ni - 1) break;
    float hs = h_arr[step];
    for (int st = 0; st < 4; ++st) {
      float cf = (st == 0) ? 0.f : ((st == 3) ? hs : 0.5f * hs);
      for (int i = tid; i < 2048; i += 256) {
        float v = sz[i];
        if (st > 0) v += cf * sk[(st - 1) * 2048 + i];
        szin[i] = (_Float16)v;
      }
      __syncthreads();
      // layer 1: [32x64] @ [64x128] -> tanh -> sa1
      {
        int nt0 = (wave & 3) * 2;
        v8f acc[2] = {vzero, vzero};
        for (int kt = 0; kt < 2; ++kt) {
          v16h a = load_a_frag_h(szin, 64, mt * 16, kt * 32, lane);
#pragma unroll
          for (int j = 0; j < 2; ++j) {
            v16h b = *(const v16h*)(sW1 + (size_t)((nt0 + j) * 2 + kt) * 512 +
                                    lane * 16);
            acc[j] = wmma16(a, b, acc[j]);
          }
        }
        int mb = mt * 16 + 8 * (lane >> 4);
#pragma unroll
        for (int j = 0; j < 2; ++j) {
          int n = (nt0 + j) * 16 + (lane & 15);
#pragma unroll
          for (int r = 0; r < 8; ++r)
            sa1[(mb + r) * 128 + n] = (_Float16)tanhf(acc[j][r] + b1[n]);
        }
      }
      __syncthreads();
      // layer 2: [32x128] @ [128x128] -> tanh -> sa2
      {
        int nt0 = (wave & 3) * 2;
        v8f acc[2] = {vzero, vzero};
        for (int kt = 0; kt < 4; ++kt) {
          v16h a = load_a_frag_h(sa1, 128, mt * 16, kt * 32, lane);
#pragma unroll
          for (int j = 0; j < 2; ++j) {
            v16h b = *(const v16h*)(sW2 + (size_t)((nt0 + j) * 4 + kt) * 512 +
                                    lane * 16);
            acc[j] = wmma16(a, b, acc[j]);
          }
        }
        int mb = mt * 16 + 8 * (lane >> 4);
#pragma unroll
        for (int j = 0; j < 2; ++j) {
          int n = (nt0 + j) * 16 + (lane & 15);
#pragma unroll
          for (int r = 0; r < 8; ++r)
            sa2[(mb + r) * 128 + n] = (_Float16)tanhf(acc[j][r] + b2[n]);
        }
      }
      __syncthreads();
      // layer 3: [32x128] @ [128x64] -> sk[st]
      {
        int nt = wave & 3;
        v8f acc = vzero;
        for (int kt = 0; kt < 4; ++kt) {
          v16h a = load_a_frag_h(sa2, 128, mt * 16, kt * 32, lane);
          v16h b =
              *(const v16h*)(sW3 + (size_t)(nt * 4 + kt) * 512 + lane * 16);
          acc = wmma16(a, b, acc);
        }
        int n = nt * 16 + (lane & 15);
        int mb = mt * 16 + 8 * (lane >> 4);
#pragma unroll
        for (int r = 0; r < 8; ++r)
          sk[st * 2048 + (mb + r) * 64 + n] = acc[r] + b3[n];
      }
      __syncthreads();
    }
    for (int i = tid; i < 2048; i += 256) {
      float zn = sz[i] + hs * (1.f / 6.f) * (sk[i] + 2.f * sk[2048 + i] +
                                             2.f * sk[4096 + i] + sk[6144 + i]);
      sz[i] = zn;
      ys[((size_t)(step + 1) * Bv + row0 + (i >> 6)) * 64 + (i & 63)] = zn;
    }
    __syncthreads();
  }
}

// ---------------------------------------------------------------------------
// Trajectory interpolation (every query time is an original timepoint, so
// unique()/argmin collapse to direct interpolation at tp/SCALE).
// ---------------------------------------------------------------------------
__device__ inline float traj_at(const float* __restrict__ ys, float tau,
                                float t0, float t1, int ni, int b, int l) {
  if (ni < 2) return ys[(size_t)b * 64 + l];
  float glast = t0 + (ni - 1) * STEPF;
  if (glast > t1) glast = t1;
  int j = (int)floorf((tau - t0) / STEPF);
  if (j < 0) j = 0;
  if (j > ni - 2) j = ni - 2;
  float tl = t0 + j * STEPF;
  float tr = (j + 1 == ni - 1) ? glast : t0 + (j + 1) * STEPF;
  float d = tr - tl;
  if (d == 0.f) d = 1.f;
  float w = (tau - tl) / d;
  float a = ys[((size_t)j * Bv + b) * 64 + l];
  float c = ys[((size_t)(j + 1) * Bv + b) * 64 + l];
  return a * (1.f - w) + c * w;
}

__global__ void k_interp(const int* __restrict__ tp, const int* __restrict__ ipar,
                         const float* __restrict__ fpar,
                         const float* __restrict__ ys, float* __restrict__ zbt) {
  int idx = blockIdx.x * blockDim.x + threadIdx.x;
  if (idx >= Bv * Tv * LATv) return;
  int l = idx & 63;
  int t = (idx >> 6) & 63;
  int b = idx >> 12;
  float tau = (float)tp[b * Tv + t] / SCALEF;
  zbt[idx] = traj_at(ys, tau, fpar[0], fpar[1], ipar[2], b, l);
}

// z_eval at last valid timepoint -> d_out tail + survival-head input (K=96 pad)
__global__ void k_zeval(const int* __restrict__ tp, const int* __restrict__ sl,
                        const float* __restrict__ age, const int* __restrict__ ipar,
                        const float* __restrict__ fpar, const float* __restrict__ ys,
                        float* __restrict__ out, float* __restrict__ sin_) {
  int idx = blockIdx.x * blockDim.x + threadIdx.x;
  if (idx >= Bv * 96) return;
  int c = idx % 96;
  int b = idx / 96;
  if (c < 64) {
    int last = sl[b] - 1;
    if (last < 0) last = 0;
    float tau = (float)tp[b * Tv + last] / SCALEF;
    float v = traj_at(ys, tau, fpar[0], fpar[1], ipar[2], b, c);
    out[1026 + b * 64 + c] = v;
    sin_[b * 96 + c] = v;
  } else if (c == 64) {
    sin_[b * 96 + c] = age[b];
  } else {
    sin_[b * 96 + c] = 0.f;
  }
}

// Masked MSE accumulation (block-reduced, one atomic per block)
__global__ void k_recon(const float* __restrict__ xhat,
                        const float* __restrict__ obs,
                        const unsigned char* __restrict__ mask,
                        const int* __restrict__ sl, float* acc) {
  __shared__ float ssq[256];
  __shared__ float scnt[256];
  int idx = blockIdx.x * blockDim.x + threadIdx.x;
  float sq = 0.f, ct = 0.f;
  if (idx < Bv * Tv * NLv) {
    int t = (idx >> 5) & 63;
    int b = idx >> 11;
    if (mask[idx] && t < sl[b]) {
      float d = xhat[idx] - obs[idx];  // xhat laid out [(b*T+t)*32 + l] == idx
      sq = d * d;
      ct = 1.f;
    }
  }
  ssq[threadIdx.x] = sq;
  scnt[threadIdx.x] = ct;
  __syncthreads();
  for (int s = 128; s > 0; s >>= 1) {
    if ((int)threadIdx.x < s) {
      ssq[threadIdx.x] += ssq[threadIdx.x + s];
      scnt[threadIdx.x] += scnt[threadIdx.x + s];
    }
    __syncthreads();
  }
  if (threadIdx.x == 0) {
    atomicAdd(acc + 0, ssq[0]);
    atomicAdd(acc + 1, scnt[0]);
  }
}

// Survival head layer 2 (N=2, plain dot products)
__global__ void k_surv2(const float* __restrict__ s1, const float* __restrict__ Ws2,
                        const float* __restrict__ bs2, float* __restrict__ out) {
  int idx = blockIdx.x * blockDim.x + threadIdx.x;
  if (idx >= Bv * 2) return;
  int b = idx >> 1, o = idx & 1;
  float s = bs2[o];
  for (int j = 0; j < 128; ++j) s += s1[b * 128 + j] * Ws2[j * 2 + o];
  out[idx] = s;
}

__global__ void k_final(const float* acc, float* out) {
  float n = acc[1];
  out[1024] = (n > 0.f) ? acc[0] / fmaxf(n, 1.f) : 0.f;
  out[1025] = -0.5f * acc[2] / (float)Bv;
}

// ---------------------------------------------------------------------------
// Host orchestration
// ---------------------------------------------------------------------------
extern "C" void kernel_launch(void* const* d_in, const int* in_sizes, int n_in,
                              void* d_out, int out_size, void* d_ws,
                              size_t ws_size, hipStream_t stream) {
  (void)in_sizes; (void)n_in; (void)out_size; (void)ws_size;
  const float* obs = (const float*)d_in[0];
  const float* age = (const float*)d_in[1];
  const float* eps = (const float*)d_in[2];
  const float* Wih = (const float*)d_in[3];
  const float* Whh = (const float*)d_in[4];
  const float* bih = (const float*)d_in[5];
  const float* bhh = (const float*)d_in[6];
  const float* Wmu = (const float*)d_in[7];
  const float* bmu = (const float*)d_in[8];
  const float* Wlv = (const float*)d_in[9];
  const float* blv = (const float*)d_in[10];
  const float* W1 = (const float*)d_in[11];
  const float* b1 = (const float*)d_in[12];
  const float* W2 = (const float*)d_in[13];
  const float* b2 = (const float*)d_in[14];
  const float* W3 = (const float*)d_in[15];
  const float* b3 = (const float*)d_in[16];
  const float* Wo1 = (const float*)d_in[17];
  const float* bo1 = (const float*)d_in[18];
  const float* Wo2 = (const float*)d_in[19];
  const float* bo2 = (const float*)d_in[20];
  const float* Ws1 = (const float*)d_in[21];
  const float* bs1 = (const float*)d_in[22];
  const float* Ws2 = (const float*)d_in[23];
  const float* bs2 = (const float*)d_in[24];
  const unsigned char* mask = (const unsigned char*)d_in[25];
  const int* tp = (const int*)d_in[26];
  const int* sl = (const int*)d_in[27];
  float* out = (float*)d_out;

  char* w = (char*)d_ws;
  size_t off = 0;
  auto take = [&](size_t bytes) {
    char* p = w + off;
    off += (bytes + 255) & ~(size_t)255;
    return p;
  };
  int* ipar = (int*)take(64);
  float* fpar = (float*)take(64);
  float* acc = (float*)take(64);
  float* h_arr = (float*)take(MAXG * 4);
  _Float16* Wih_p = (_Float16*)take((size_t)24 * 3 * 512 * 2);
  _Float16* Whh_p = (_Float16*)take((size_t)24 * 4 * 512 * 2);
  _Float16* Wmu_p = (_Float16*)take((size_t)4 * 4 * 512 * 2);
  _Float16* Wlv_p = (_Float16*)take((size_t)4 * 4 * 512 * 2);
  _Float16* W1_p = (_Float16*)take((size_t)8 * 2 * 512 * 2);
  _Float16* W2_p = (_Float16*)take((size_t)8 * 4 * 512 * 2);
  _Float16* W3_p = (_Float16*)take((size_t)4 * 4 * 512 * 2);
  _Float16* Wo1_p = (_Float16*)take((size_t)8 * 2 * 512 * 2);
  _Float16* Wo2_p = (_Float16*)take((size_t)2 * 4 * 512 * 2);
  _Float16* Ws1_p = (_Float16*)take((size_t)8 * 3 * 512 * 2);
  float* Xrev = (float*)take((size_t)Tv * Bv * 96 * 4);
  float* gi_all = (float*)take((size_t)Tv * Bv * 384 * 4);
  float* hfin = (float*)take((size_t)Bv * 128 * 4);
  float* mu = (float*)take((size_t)Bv * 64 * 4);
  float* lv = (float*)take((size_t)Bv * 64 * 4);
  float* ys = (float*)take((size_t)MAXG * Bv * 64 * 4);
  float* zbt = (float*)take((size_t)Bv * Tv * 64 * 4);
  float* dec1 = (float*)take((size_t)Bv * Tv * 128 * 4);
  float* xh = (float*)take((size_t)Bv * Tv * 32 * 4);
  float* sin_ = (float*)take((size_t)Bv * 96 * 4);
  float* s1 = (float*)take((size_t)Bv * 128 * 4);

  auto pack = [&](const float* Wm, int K, int N, _Float16* dst) {
    int KT = (K + 31) / 32, NT = N / 16;
    int threads = NT * KT * 32;
    k_pack<<<(threads + 255) / 256, 256, 0, stream>>>(Wm, K, N, KT, NT, dst);
  };
  auto gemm4 = [&](const float* A, int lda, const _Float16* Bp,
                   const float* bias, float* C, int ldc, int M, int Kp, int N,
                   int act) {
    int groups = (M / 16) * (N / 16 / 4);
    int blocks = (groups * 32 + 255) / 256;
    k_gemm<4><<<blocks, 256, 0, stream>>>(A, lda, Bp, bias, C, ldc, M, Kp, N,
                                          act);
  };
  auto gemm2 = [&](const float* A, int lda, const _Float16* Bp,
                   const float* bias, float* C, int ldc, int M, int Kp, int N,
                   int act) {
    int groups = (M / 16) * (N / 16 / 2);
    int blocks = (groups * 32 + 255) / 256;
    k_gemm<2><<<blocks, 256, 0, stream>>>(A, lda, Bp, bias, C, ldc, M, Kp, N,
                                          act);
  };

  // setup
  k_init<<<1, 64, 0, stream>>>(ipar, acc);
  k_minmax<<<(Bv * Tv + 255) / 256, 256, 0, stream>>>(tp, ipar);
  k_grid<<<1, 128, 0, stream>>>(ipar, fpar, h_arr);

  // pack weights into WMMA B-fragment layout
  pack(Wih, 65, 384, Wih_p);
  pack(Whh, 128, 384, Whh_p);
  pack(Wmu, 128, 64, Wmu_p);
  pack(Wlv, 128, 64, Wlv_p);
  pack(W1, 64, 128, W1_p);
  pack(W2, 128, 128, W2_p);
  pack(W3, 128, 64, W3_p);
  pack(Wo1, 64, 128, Wo1_p);
  pack(Wo2, 128, 32, Wo2_p);
  pack(Ws1, 65, 128, Ws1_p);

  // GRU input path
  k_build_x<<<(Tv * Bv * 96 + 255) / 256, 256, 0, stream>>>(obs, mask, tp, Xrev);
  gemm4(Xrev, 96, Wih_p, bih, gi_all, 384, Tv * Bv, 96, 384, 0);  // gi, all t

  const int GRU_SMEM = 49152 * 2 + 32 * 128 * 2 + 32 * 384 * 4;  // 155648 B
  k_gru<<<16, 256, GRU_SMEM, stream>>>(gi_all, Whh_p, bhh, hfin);

  // heads + reparameterization
  gemm4(hfin, 128, Wmu_p, bmu, mu, 64, Bv, 128, 64, 0);
  gemm4(hfin, 128, Wlv_p, blv, lv, 64, Bv, 128, 64, 0);
  k_z0<<<(Bv * 64 + 255) / 256, 256, 0, stream>>>(mu, lv, eps, ys, acc);

  // ODE integration (persistent, all RK4 steps in one launch)
  const int ODE_SMEM =
      (8192 + 16384 + 8192 + 2048 + 4096 + 4096) * 2 + (2048 + 8192) * 4;
  k_ode<<<16, 256, ODE_SMEM, stream>>>(W1_p, W2_p, W3_p, b1, b2, b3, ipar,
                                       h_arr, ys);

  // trajectory -> per-(b,t) latent and z_eval
  k_interp<<<(Bv * Tv * 64 + 255) / 256, 256, 0, stream>>>(tp, ipar, fpar, ys,
                                                           zbt);
  k_zeval<<<(Bv * 96 + 255) / 256, 256, 0, stream>>>(tp, sl, age, ipar, fpar,
                                                     ys, out, sin_);

  // decoder + recon loss
  gemm4(zbt, 64, Wo1_p, bo1, dec1, 128, Bv * Tv, 64, 128, 2);
  gemm2(dec1, 128, Wo2_p, bo2, xh, 32, Bv * Tv, 128, 32, 0);
  k_recon<<<(Bv * Tv * NLv + 255) / 256, 256, 0, stream>>>(xh, obs, mask, sl,
                                                           acc);

  // survival head
  gemm4(sin_, 96, Ws1_p, bs1, s1, 128, Bv, 96, 128, 2);
  k_surv2<<<(Bv * 2 + 255) / 256, 256, 0, stream>>>(s1, Ws2, bs2, out);

  // scalars
  k_final<<<1, 1, 0, stream>>>(acc, out);
}
